// KAN_Encode_34351148433544
// MI455X (gfx1250) — matrix operand order repaired and checked by reading, not compiled, not run on hardware
//
#include <hip/hip_runtime.h>
#include <hip/hip_bf16.h>

// ---------------------------------------------------------------------------
// Types for CDNA5 WMMA
// ---------------------------------------------------------------------------
typedef __bf16          v16bf __attribute__((ext_vector_type(16)));
typedef __bf16          v8bf  __attribute__((ext_vector_type(8)));
typedef float           v8f   __attribute__((ext_vector_type(8)));
typedef unsigned short  us8   __attribute__((ext_vector_type(8)));

// float -> bf16 (round to nearest even), stored as raw u16
static __device__ __forceinline__ unsigned short f2bf(float f) {
    unsigned u = __builtin_bit_cast(unsigned, f);
    u += 0x7FFFu + ((u >> 16) & 1u);
    return (unsigned short)(u >> 16);
}

// concat two 8-element raw-bf16 chunks into a 16-element WMMA fragment
static __device__ __forceinline__ v16bf frag_cat(us8 lo, us8 hi) {
    v8bf l = __builtin_bit_cast(v8bf, lo);
    v8bf h = __builtin_bit_cast(v8bf, hi);
    return __builtin_shufflevector(l, h, 0,1,2,3,4,5,6,7,8,9,10,11,12,13,14,15);
}

// ---------------------------------------------------------------------------
// Cubic B-spline bases (grid_size=5, order=3, range [-1,1] -> knots step 0.4)
// Cox–de Boor, fully unrolled; yields 8 basis values.
// ---------------------------------------------------------------------------
static __device__ __forceinline__ void bsplines8(float x, float* __restrict__ out) {
    const float t[12] = {-2.2f,-1.8f,-1.4f,-1.0f,-0.6f,-0.2f,
                          0.2f, 0.6f, 1.0f, 1.4f, 1.8f, 2.2f};
    float b[11];
#pragma unroll
    for (int i = 0; i < 11; ++i)
        b[i] = (x >= t[i] && x < t[i + 1]) ? 1.0f : 0.0f;
    {
        const float inv = 1.0f / 0.4f;
#pragma unroll
        for (int i = 0; i < 10; ++i)
            b[i] = (x - t[i]) * inv * b[i] + (t[i + 2] - x) * inv * b[i + 1];
    }
    {
        const float inv = 1.0f / 0.8f;
#pragma unroll
        for (int i = 0; i < 9; ++i)
            b[i] = (x - t[i]) * inv * b[i] + (t[i + 3] - x) * inv * b[i + 1];
    }
    {
        const float inv = 1.0f / 1.2f;
#pragma unroll
        for (int i = 0; i < 8; ++i)
            b[i] = (x - t[i]) * inv * b[i] + (t[i + 4] - x) * inv * b[i + 1];
    }
#pragma unroll
    for (int i = 0; i < 8; ++i) out[i] = b[i];
}

// ---------------------------------------------------------------------------
// Build concatenated KAN activation matrix: ACT(B, 9*D) bf16
//   cols [0,D)        : silu(h)
//   cols [D + 8i, +8) : bases(h[:,i])
// ---------------------------------------------------------------------------
__global__ void kan_act_kernel(const float* __restrict__ H,
                               unsigned short* __restrict__ ACT,
                               int D, int total) {
    int t = blockIdx.x * blockDim.x + threadIdx.x;
    if (t >= total) return;
    int b = t / D, i = t - b * D;
    size_t ld = (size_t)D * 9;
    float x = H[(size_t)b * D + i];
    float s = x / (1.0f + __expf(-x));                 // SiLU
    ACT[(size_t)b * ld + i] = f2bf(s);
    float bs[8];
    bsplines8(x, bs);
    us8 v;
#pragma unroll
    for (int e = 0; e < 8; ++e) v[e] = f2bf(bs[e]);
    *reinterpret_cast<us8*>(ACT + (size_t)b * ld + D + (size_t)i * 8) = v;  // 16B store
}

// ---------------------------------------------------------------------------
// Fuse base weight + scaled spline weight into W(O, 9*D) bf16
// ---------------------------------------------------------------------------
__global__ void kan_prep_w_kernel(const float* __restrict__ bw,
                                  const float* __restrict__ sw,
                                  const float* __restrict__ ss,
                                  unsigned short* __restrict__ W,
                                  int D, int total) {
    int t = blockIdx.x * blockDim.x + threadIdx.x;
    if (t >= total) return;
    int o = t / D, i = t - o * D;
    size_t ld = (size_t)D * 9;
    W[(size_t)o * ld + i] = f2bf(bw[t]);
    float s = ss[t];
    const float* sp = sw + (size_t)t * 8;
    us8 v;
#pragma unroll
    for (int e = 0; e < 8; ++e) v[e] = f2bf(sp[e] * s);
    *reinterpret_cast<us8*>(W + (size_t)o * ld + D + (size_t)i * 8) = v;    // 16B store
}

// ---------------------------------------------------------------------------
// f32 -> bf16 cast (optional ReLU)
// ---------------------------------------------------------------------------
__global__ void cast_bf16_kernel(const float* __restrict__ in,
                                 unsigned short* __restrict__ out,
                                 int n, int relu) {
    int t = blockIdx.x * blockDim.x + threadIdx.x;
    if (t >= n) return;
    float v = in[t];
    if (relu) v = v > 0.0f ? v : 0.0f;
    out[t] = f2bf(v);
}

// ---------------------------------------------------------------------------
// bf16 WMMA GEMM:  C(M,N) = A(M,K) * Bw(N,K)^T [+ bias]
// Block: 256 threads = 8 waves; block tile 128x256; wave tile 64x64 (4x4 WMMA).
// Waves arranged 2 (M) x 4 (N).  Inner loop: 16 b128 loads : 16 WMMA.
// Fragment layouts per CDNA5 ISA §7.12.2 (wave32):
//   A: lane row = lane%16; v16 elems 0..7 -> K = g*8..g*8+7, elems 8..15 -> K+16
//   B: lane col = lane%16; v16 elems 0..7 -> K = g*8..g*8+7, elems 8..15 -> K+16   (g = lane/16)
// ---------------------------------------------------------------------------
__global__ __launch_bounds__(256)
void gemm_bf16_kernel(const unsigned short* __restrict__ A,
                      const unsigned short* __restrict__ Bw,
                      float* __restrict__ C,
                      const float* __restrict__ bias,
                      int K, int lda, int ldb, int ldc) {
    const int lane = threadIdx.x & 31;
    const int wave = threadIdx.x >> 5;
    const int g    = lane >> 4;
    const int ln   = lane & 15;
    const int bm   = blockIdx.y * 128 + (wave >> 2) * 64;   // wave row base
    const int bn   = blockIdx.x * 256 + (wave & 3) * 64;    // wave col base

    const unsigned short* ap[4];
    const unsigned short* bp[4];
#pragma unroll
    for (int i = 0; i < 4; ++i)
        ap[i] = A + (size_t)(bm + i * 16 + ln) * lda + g * 8;
#pragma unroll
    for (int j = 0; j < 4; ++j)
        bp[j] = Bw + (size_t)(bn + j * 16 + ln) * ldb + g * 8;

    const v8f zacc = {0.f, 0.f, 0.f, 0.f, 0.f, 0.f, 0.f, 0.f};
    v8f acc[4][4];
#pragma unroll
    for (int i = 0; i < 4; ++i)
#pragma unroll
        for (int j = 0; j < 4; ++j) acc[i][j] = zacc;

    for (int k = 0; k < K; k += 32) {
        v16bf af[4], bf[4];
#pragma unroll
        for (int i = 0; i < 4; ++i)
            af[i] = frag_cat(*reinterpret_cast<const us8*>(ap[i] + k),
                             *reinterpret_cast<const us8*>(ap[i] + k + 16));
#pragma unroll
        for (int j = 0; j < 4; ++j)
            bf[j] = frag_cat(*reinterpret_cast<const us8*>(bp[j] + k),
                             *reinterpret_cast<const us8*>(bp[j] + k + 16));
#pragma unroll
        for (int i = 0; i < 4; ++i)
#pragma unroll
            for (int j = 0; j < 4; ++j)
                acc[i][j] = __builtin_amdgcn_wmma_f32_16x16x32_bf16(
                    false, af[i], false, bf[j], (short)0, acc[i][j], false, false);
    }

    // C/D layout: v8f elem e -> row = base + i*16 + g*8 + e, col = base + j*16 + ln
#pragma unroll
    for (int j = 0; j < 4; ++j) {
        const int col = bn + j * 16 + ln;
        const float bi = bias ? bias[col] : 0.0f;
#pragma unroll
        for (int i = 0; i < 4; ++i) {
#pragma unroll
            for (int e = 0; e < 8; ++e) {
                const int row = bm + i * 16 + g * 8 + e;
                C[(size_t)row * ldc + col] = acc[i][j][e] + bi;
            }
        }
    }
}

// ---------------------------------------------------------------------------
// Orchestration
// ---------------------------------------------------------------------------
extern "C" void kernel_launch(void* const* d_in, const int* in_sizes, int n_in,
                              void* d_out, int out_size, void* d_ws, size_t ws_size,
                              hipStream_t stream) {
    (void)in_sizes; (void)n_in; (void)out_size; (void)ws_size;

    const float* x   = (const float*)d_in[0];
    const float* bw0 = (const float*)d_in[1];
    const float* sw0 = (const float*)d_in[2];
    const float* ss0 = (const float*)d_in[3];
    const float* bw1 = (const float*)d_in[4];
    const float* sw1 = (const float*)d_in[5];
    const float* ss1 = (const float*)d_in[6];
    const float* dw  = (const float*)d_in[7];
    const float* db  = (const float*)d_in[8];
    float* out = (float*)d_out;

    const int B = 4096, D0 = 1024, H0 = 2048, H1 = 1024, L = 512;

    // Workspace layout (stream-ordered reuse; peak ~228 MB)
    char* ws = (char*)d_ws;
    unsigned short* act0 = (unsigned short*)(ws + 0);           //  75,497,472 B (B x 9*D0)
    unsigned short* w0   = (unsigned short*)(ws + 75497472);    //  37,748,736 B (H0 x 9*D0)
    float*          h0   = (float*)        (ws + 150994944);    //  33,554,432 B (B x H0)
    unsigned short* act1 = (unsigned short*)(ws + 0);           // 150,994,944 B (B x 9*H0) reuses act0+w0
    unsigned short* w1   = (unsigned short*)(ws + 184549376);   //  37,748,736 B (H1 x 9*H0)
    float*          h1   = (float*)        (ws + 222298112);    //  16,777,216 B (B x H1)
    unsigned short* a2   = (unsigned short*)(ws + 0);           //   8,388,608 B (B x H1) reuses act1
    unsigned short* w2   = (unsigned short*)(ws + 8388608);     //   1,048,576 B (L x H1)

    // ---- Layer 0 ----
    kan_prep_w_kernel<<<(H0 * D0) / 256, 256, 0, stream>>>(bw0, sw0, ss0, w0, D0, H0 * D0);
    kan_act_kernel<<<(B * D0) / 256, 256, 0, stream>>>(x, act0, D0, B * D0);
    gemm_bf16_kernel<<<dim3(H0 / 256, B / 128), 256, 0, stream>>>(
        act0, w0, h0, nullptr, 9 * D0, 9 * D0, 9 * D0, H0);

    // ---- Layer 1 ----
    kan_prep_w_kernel<<<(H1 * H0) / 256, 256, 0, stream>>>(bw1, sw1, ss1, w1, H0, H1 * H0);
    kan_act_kernel<<<(B * H0) / 256, 256, 0, stream>>>(h0, act1, H0, B * H0);
    gemm_bf16_kernel<<<dim3(H1 / 256, B / 128), 256, 0, stream>>>(
        act1, w1, h1, nullptr, 9 * H0, 9 * H0, 9 * H0, H1);

    // ---- ReLU + dense head ----
    cast_bf16_kernel<<<(B * H1) / 256, 256, 0, stream>>>(h1, a2, B * H1, 1);
    cast_bf16_kernel<<<(L * H1) / 256, 256, 0, stream>>>(dw, w2, L * H1, 0);
    gemm_bf16_kernel<<<dim3(L / 256, B / 128), 256, 0, stream>>>(
        a2, w2, out, db, H1, H1, H1, L);
}
